// TFL_34170759807005
// MI455X (gfx1250) — matrix operand order repaired and checked by reading, not compiled, not run on hardware
//
#include <hip/hip_runtime.h>
#include <hip/hip_bf16.h>

#define DIMC   192
#define HEADS  6
#define NTOK   64
#define NWIN   1024
#define HD     32

typedef _Float16 h16v __attribute__((ext_vector_type(16)));
typedef _Float16 h8v  __attribute__((ext_vector_type(8)));
typedef _Float16 h4v  __attribute__((ext_vector_type(4)));
typedef float    f8v  __attribute__((ext_vector_type(8)));

// ---------------------------------------------------------------------------
// WMMA fragment helpers (wave32, V_WMMA_F32_16X16X32_F16)
// A (16x32 f16): lane = 16*hi + m ; halfs 0-7 = K[hi*8 .. hi*8+7],
//                halfs 8-15 = K[16+hi*8 .. 16+hi*8+7]   (ISA 7.12.2 table)
// B (32x16 f16): lane = 16*hi + n ; halfs 0-7 = K[hi*8 .. hi*8+7],
//                halfs 8-15 = K[16+hi*8 .. 16+hi*8+7]   (same as A)
// C/D (16x16 f32): elem j -> row (j + hi*8), col (lane & 15)
// ---------------------------------------------------------------------------
__device__ __forceinline__ h16v frag_a(const _Float16* p) {
    int hi = (threadIdx.x >> 4) & 1;
    h16v r;
    *(h8v*)&r       = *(const h8v*)(p + hi * 8);
    *((h8v*)&r + 1) = *(const h8v*)(p + 16 + hi * 8);
    return r;
}
__device__ __forceinline__ h16v frag_b(const _Float16* p) {
    int hi = (threadIdx.x >> 4) & 1;
    h16v r;
    *(h8v*)&r       = *(const h8v*)(p + hi * 8);
    *((h8v*)&r + 1) = *(const h8v*)(p + 16 + hi * 8);
    return r;
}
__device__ __forceinline__ f8v wmma32(h16v a, h16v b, f8v c) {
    return __builtin_amdgcn_wmma_f32_16x16x32_f16(false, a, false, b,
                                                  (short)0, c, false, false);
}

// One weight column-tile (16 output features) vs all 4 row-tiles of a 64-row
// activation panel: each B k-fragment is loaded from global exactly once and
// reused by 4 WMMAs (4 independent accumulation chains).
template <int NK>
__device__ __forceinline__ void gemm_col(const _Float16* Abase, int lda,
                                         const _Float16* brow, f8v acc[4], int ln) {
#pragma unroll
    for (int i = 0; i < 4; ++i)
        acc[i] = (f8v){0.f, 0.f, 0.f, 0.f, 0.f, 0.f, 0.f, 0.f};
#pragma unroll
    for (int k = 0; k < NK; ++k) {
        h16v b = frag_b(brow + k * 32);
#pragma unroll
        for (int tm = 0; tm < 4; ++tm) {
            h16v a = frag_a(Abase + (tm * 16 + ln) * lda + k * 32);
            acc[tm] = wmma32(a, b, acc[tm]);
        }
    }
}

// ---------------------------------------------------------------------------
// Per-head attention: S = Q Kt (wmma) -> +bias +mask softmax -> O = P V (wmma)
// Q, Kb: 64x192 row-major f16 (head h in cols [32h,32h+32))
// VT   : 192x64 f16 (channel-major; row c holds v[:,c])
// S    : 64x64 f32 scratch ; P : 64x64 f16 scratch
// OUTb : 64x384 f16 concat buffer, this pass writes cols [outcol, outcol+192)
// ---------------------------------------------------------------------------
__device__ __forceinline__ void attention_pass(
    int w, const _Float16* Q, const _Float16* Kb, const _Float16* VT,
    float* S, _Float16* P, _Float16* OUTb, int outcol,
    const float* __restrict__ table, const float* __restrict__ mask,
    const int* __restrict__ rel_idx)
{
    const int tid = threadIdx.x;
    const int ln = tid & 15, hi = (tid >> 4) & 1;
    const int wv = __builtin_amdgcn_readfirstlane(tid >> 5);   // scalar wave id
    const float* mrow_base = mask + (size_t)(w % NWIN) * NTOK * NTOK;

    for (int h = 0; h < HEADS; ++h) {
        // ---- scores: wave owns column tn = wv&3, rows tm0 and tm0+2;
        //      K-row B-fragment loaded once, reused by both WMMAs ----
        {
            int tn = wv & 3, tm0 = wv >> 2;      // scalar
            h16v b = frag_b(Kb + (tn * 16 + ln) * DIMC + h * HD);
#pragma unroll
            for (int t0 = 0; t0 < 2; ++t0) {
                int tm = tm0 + t0 * 2;
                f8v acc = {0.f, 0.f, 0.f, 0.f, 0.f, 0.f, 0.f, 0.f};
                h16v a = frag_a(Q + (tm * 16 + ln) * DIMC + h * HD);
                acc = wmma32(a, b, acc);
#pragma unroll
                for (int j = 0; j < 8; ++j)
                    S[(tm * 16 + hi * 8 + j) * NTOK + tn * 16 + ln] = acc[j];
            }
        }
        __syncthreads();

        // ---- softmax: 4 lanes per row, 16 elems/lane in registers,
        //      quad reductions via __shfl_xor (lane-permute) ----
        {
            const int r  = tid >> 2;             // row 0..63
            const int q0 = (tid & 3) * 16;       // quarter offset
            const float* srow = S + r * NTOK + q0;
            const float* mrow = mrow_base + r * NTOK + q0;
            const int*   ridx = rel_idx + r * NTOK + q0;
            float vals[16];
            float mx = -3.40282347e38f;
#pragma unroll
            for (int j = 0; j < 16; ++j) {
                float v = srow[j] + table[ridx[j] * HEADS + h] + mrow[j];
                vals[j] = v;
                mx = fmaxf(mx, v);
            }
            mx = fmaxf(mx, __shfl_xor(mx, 1));
            mx = fmaxf(mx, __shfl_xor(mx, 2));
            float sum = 0.f;
#pragma unroll
            for (int j = 0; j < 16; ++j) {
                float e = __expf(vals[j] - mx);
                vals[j] = e;
                sum += e;
            }
            sum += __shfl_xor(sum, 1);
            sum += __shfl_xor(sum, 2);
            float inv = 1.f / sum;
            _Float16* prow = P + r * NTOK + q0;
#pragma unroll
            for (int j = 0; j < 16; ++j)
                prow[j] = (_Float16)(vals[j] * inv);
        }
        __syncthreads();

        // ---- O = P @ V : 4x2 tiles of 16x16, K = 64 (2 wmma each) ----
        {
            int tm = wv >> 1, td = wv & 1;       // scalar
            f8v acc = {0.f, 0.f, 0.f, 0.f, 0.f, 0.f, 0.f, 0.f};
#pragma unroll
            for (int nc = 0; nc < 2; ++nc) {
                h16v a = frag_a(P  + (tm * 16 + ln) * NTOK + nc * 32);
                h16v b = frag_b(VT + (h * HD + td * 16 + ln) * NTOK + nc * 32);
                acc = wmma32(a, b, acc);
            }
#pragma unroll
            for (int j = 0; j < 8; ++j)
                OUTb[(tm * 16 + hi * 8 + j) * (2 * DIMC) +
                     outcol + h * HD + td * 16 + ln] = (_Float16)acc[j];
        }
        __syncthreads();   // S/P reused by next head
    }
}

// ---------------------------------------------------------------------------
// Fused per-window kernel. One workgroup (8 wave32) = one window.
// LDS layout (144 KB):
//   [0,      24576) : XH/YH (64x192 f16)  <->  S (64x64 f32) + P (64x64 f16)
//   [24576,  49152) : Q  (64x192 f16)  -> reused as M1
//   [49152,  73728) : K  / Ky          -> reused as M2
//   [73728,  98304) : VT / VyT (192x64 f16, transposed)
//   [98304, 147456) : OUT concat (64x384 f16)
// ---------------------------------------------------------------------------
__global__ __launch_bounds__(256) void fused_window_attn(
    const float* __restrict__ x,     const float* __restrict__ y,
    const float* __restrict__ mask_x, const float* __restrict__ mask_y,
    const float* __restrict__ qkv_b, const float* __restrict__ kv_b,
    const float* __restrict__ rpb_x, const float* __restrict__ rpb_y,
    const float* __restrict__ m1_b,  const float* __restrict__ m2_b,
    const float* __restrict__ proj_b, const int* __restrict__ rel_idx,
    const _Float16* __restrict__ wqkv, const _Float16* __restrict__ wkv,
    const _Float16* __restrict__ wm1,  const _Float16* __restrict__ wm2,
    const _Float16* __restrict__ wpr,  float* __restrict__ out)
{
    __shared__ __align__(16) unsigned char smem[147456];
    _Float16* R0h  = (_Float16*)smem;              // XH / YH
    float*    S    = (float*)smem;                 // scores f32
    _Float16* P    = (_Float16*)(smem + 16384);    // probs f16
    _Float16* Q    = (_Float16*)(smem + 24576);
    _Float16* Kb   = (_Float16*)(smem + 49152);
    _Float16* VT   = (_Float16*)(smem + 73728);
    _Float16* OUTb = (_Float16*)(smem + 98304);

    const int w   = blockIdx.x;
    const int tid = threadIdx.x;
    const int ln  = tid & 15, hi = (tid >> 4) & 1;
    const int wv  = __builtin_amdgcn_readfirstlane(tid >> 5);  // scalar wave id
    const float scale = 0.17677669529663687f;      // 32^-0.5

    // warm L2/WGP$ with weight streams (global_prefetch_b8)
    __builtin_prefetch(wqkv + tid * 64, 0, 1);
    __builtin_prefetch(wkv  + tid * 32, 0, 1);
    __builtin_prefetch(wm1  + tid * 32, 0, 1);

    // ---- stage x -> f16 LDS ----
    {
        const float4* xs = (const float4*)(x + (size_t)w * NTOK * DIMC);
        for (int i = tid; i < NTOK * DIMC / 4; i += 256) {
            float4 v = xs[i];
            h4v hh = {(_Float16)v.x, (_Float16)v.y, (_Float16)v.z, (_Float16)v.w};
            *(h4v*)(R0h + i * 4) = hh;
        }
    }
    __syncthreads();

    // ---- qkv GEMM: (64x192)x(192x576) -> Q,K,VT  (36 col-tiles, K=192) ----
    for (int tc = wv; tc < 36; tc += 8) {
        const _Float16* brow = wqkv + (tc * 16 + ln) * DIMC;
        f8v acc[4];
        gemm_col<6>(R0h, DIMC, brow, acc, ln);
        int col = tc * 16;                              // scalar
        float bias = qkv_b[col + ln];
        if (col < 192) {                                // scalar branch -> Q
#pragma unroll
            for (int tm = 0; tm < 4; ++tm)
#pragma unroll
                for (int j = 0; j < 8; ++j)
                    Q[(tm * 16 + hi * 8 + j) * DIMC + col + ln] =
                        (_Float16)((acc[tm][j] + bias) * scale);
        } else if (col < 384) {                         // -> K
#pragma unroll
            for (int tm = 0; tm < 4; ++tm)
#pragma unroll
                for (int j = 0; j < 8; ++j)
                    Kb[(tm * 16 + hi * 8 + j) * DIMC + (col - 192) + ln] =
                        (_Float16)(acc[tm][j] + bias);
        } else {                                        // -> V transposed
#pragma unroll
            for (int tm = 0; tm < 4; ++tm)
#pragma unroll
                for (int j = 0; j < 8; ++j)
                    VT[((col - 384) + ln) * NTOK + (tm * 16 + hi * 8 + j)] =
                        (_Float16)(acc[tm][j] + bias);
        }
    }
    __syncthreads();

    // ---- self attention -> OUT[:, 0:192] ----
    attention_pass(w, Q, Kb, VT, S, P, OUTb, 0, rpb_x, mask_x, rel_idx);

    // ---- stage y -> f16 LDS (reuse R0) ----
    {
        const float4* ys = (const float4*)(y + (size_t)w * NTOK * DIMC);
        for (int i = tid; i < NTOK * DIMC / 4; i += 256) {
            float4 v = ys[i];
            h4v hh = {(_Float16)v.x, (_Float16)v.y, (_Float16)v.z, (_Float16)v.w};
            *(h4v*)(R0h + i * 4) = hh;
        }
    }
    __syncthreads();

    // ---- kv GEMM: (64x192)x(192x384) -> Ky,VyT (24 col-tiles) ----
    for (int tc = wv; tc < 24; tc += 8) {
        const _Float16* brow = wkv + (tc * 16 + ln) * DIMC;
        f8v acc[4];
        gemm_col<6>(R0h, DIMC, brow, acc, ln);
        int col = tc * 16;                              // scalar
        float bias = kv_b[col + ln];
        if (col < 192) {                                // scalar branch -> Ky
#pragma unroll
            for (int tm = 0; tm < 4; ++tm)
#pragma unroll
                for (int j = 0; j < 8; ++j)
                    Kb[(tm * 16 + hi * 8 + j) * DIMC + col + ln] =
                        (_Float16)(acc[tm][j] + bias);
        } else {                                        // -> Vy transposed
#pragma unroll
            for (int tm = 0; tm < 4; ++tm)
#pragma unroll
                for (int j = 0; j < 8; ++j)
                    VT[((col - 192) + ln) * NTOK + (tm * 16 + hi * 8 + j)] =
                        (_Float16)(acc[tm][j] + bias);
        }
    }
    __syncthreads();

    // ---- cross attention -> OUT[:, 192:384] ----
    attention_pass(w, Q, Kb, VT, S, P, OUTb, 192, rpb_y, mask_y, rel_idx);

    // ---- merge1: (64x384)x(384x192) + bias + leaky_relu(0.2) -> M1 ----
    _Float16* M1 = Q;   // Q dead
    for (int tc = wv; tc < 12; tc += 8) {
        const _Float16* brow = wm1 + (tc * 16 + ln) * 384;
        f8v acc[4];
        gemm_col<12>(OUTb, 384, brow, acc, ln);
        float bias = m1_b[tc * 16 + ln];
#pragma unroll
        for (int tm = 0; tm < 4; ++tm)
#pragma unroll
            for (int j = 0; j < 8; ++j) {
                float v = acc[tm][j] + bias;
                v = v > 0.f ? v : 0.2f * v;
                M1[(tm * 16 + hi * 8 + j) * DIMC + tc * 16 + ln] = (_Float16)v;
            }
    }
    __syncthreads();

    // ---- merge2: (64x192)x(192x192) + bias -> M2 ----
    _Float16* M2 = Kb;  // Kb dead
    for (int tc = wv; tc < 12; tc += 8) {
        const _Float16* brow = wm2 + (tc * 16 + ln) * DIMC;
        f8v acc[4];
        gemm_col<6>(M1, DIMC, brow, acc, ln);
        float bias = m2_b[tc * 16 + ln];
#pragma unroll
        for (int tm = 0; tm < 4; ++tm)
#pragma unroll
            for (int j = 0; j < 8; ++j)
                M2[(tm * 16 + hi * 8 + j) * DIMC + tc * 16 + ln] =
                    (_Float16)(acc[tm][j] + bias);
    }
    __syncthreads();

    // ---- proj: (64x192)x(192x192) + bias -> global f32 ----
    for (int tc = wv; tc < 12; tc += 8) {
        const _Float16* brow = wpr + (tc * 16 + ln) * DIMC;
        f8v acc[4];
        gemm_col<6>(M2, DIMC, brow, acc, ln);
        float bias = proj_b[tc * 16 + ln];
        float* orow = out + (size_t)w * NTOK * DIMC;
#pragma unroll
        for (int tm = 0; tm < 4; ++tm)
#pragma unroll
            for (int j = 0; j < 8; ++j)
                orow[(tm * 16 + hi * 8 + j) * DIMC + tc * 16 + ln] =
                    acc[tm][j] + bias;
    }
}

// ---------------------------------------------------------------------------
// f32 -> f16 weight pre-conversion (weights then live in 192MB L2)
// ---------------------------------------------------------------------------
__global__ __launch_bounds__(256) void cvt_f32_f16(const float* __restrict__ src,
                                                   _Float16* __restrict__ dst, int n) {
    int i = blockIdx.x * 256 + threadIdx.x;
    if (i < n) dst[i] = (_Float16)src[i];
}

extern "C" void kernel_launch(void* const* d_in, const int* in_sizes, int n_in,
                              void* d_out, int out_size, void* d_ws, size_t ws_size,
                              hipStream_t stream) {
    const float* x      = (const float*)d_in[0];
    const float* y      = (const float*)d_in[1];
    const float* mask_x = (const float*)d_in[2];
    const float* mask_y = (const float*)d_in[3];
    const float* qkv_w  = (const float*)d_in[4];
    const float* qkv_b  = (const float*)d_in[5];
    const float* kv_w   = (const float*)d_in[6];
    const float* kv_b   = (const float*)d_in[7];
    const float* rpb_x  = (const float*)d_in[8];
    const float* rpb_y  = (const float*)d_in[9];
    const float* m1_w   = (const float*)d_in[10];
    const float* m1_b   = (const float*)d_in[11];
    const float* m2_w   = (const float*)d_in[12];
    const float* m2_b   = (const float*)d_in[13];
    const float* pr_w   = (const float*)d_in[14];
    const float* pr_b   = (const float*)d_in[15];
    const int*   rel_idx = (const int*)d_in[16];
    float* out = (float*)d_out;

    _Float16* ws  = (_Float16*)d_ws;
    _Float16* wqkv = ws;                          // 576*192
    _Float16* wkv  = wqkv + 576 * 192;            // 384*192
    _Float16* wm1  = wkv  + 384 * 192;            // 192*384
    _Float16* wm2  = wm1  + 192 * 384;            // 192*192
    _Float16* wpr  = wm2  + 192 * 192;            // 192*192

    cvt_f32_f16<<<(576 * 192 + 255) / 256, 256, 0, stream>>>(qkv_w, wqkv, 576 * 192);
    cvt_f32_f16<<<(384 * 192 + 255) / 256, 256, 0, stream>>>(kv_w,  wkv,  384 * 192);
    cvt_f32_f16<<<(192 * 384 + 255) / 256, 256, 0, stream>>>(m1_w,  wm1,  192 * 384);
    cvt_f32_f16<<<(192 * 192 + 255) / 256, 256, 0, stream>>>(m2_w,  wm2,  192 * 192);
    cvt_f32_f16<<<(192 * 192 + 255) / 256, 256, 0, stream>>>(pr_w,  wpr,  192 * 192);

    fused_window_attn<<<2048, 256, 0, stream>>>(
        x, y, mask_x, mask_y, qkv_b, kv_b, rpb_x, rpb_y,
        m1_b, m2_b, pr_b, rel_idx, wqkv, wkv, wm1, wm2, wpr, out);
}
